// CausalAttention_69836168233253
// MI455X (gfx1250) — matrix operand order repaired and checked by reading, not compiled, not run on hardware
//
#include <hip/hip_runtime.h>

// ---------- types ----------
typedef __attribute__((ext_vector_type(16))) __bf16          v16bf;
typedef __attribute__((ext_vector_type(8)))  __bf16          bf16x8;
typedef __attribute__((ext_vector_type(8)))  float           v8f;
typedef __attribute__((ext_vector_type(4)))  float           f32x4;
typedef __attribute__((ext_vector_type(4)))  unsigned short  u16x4;
typedef __attribute__((ext_vector_type(8)))  unsigned short  u16x8;
typedef __attribute__((ext_vector_type(4)))  unsigned int    u32x4;
typedef __attribute__((ext_vector_type(8)))  int             i32x8;
typedef __attribute__((ext_vector_type(4)))  int             i32x4;

#define S_LEN   2048
#define D_DIM   1024
#define BATCH   4
#define NROWS   (BATCH * S_LEN)      // 8192
#define LDSB_PITCH 72                // 64 halves + 8 halves pad (bank-conflict-free)

// ---------- helpers ----------
__device__ __forceinline__ unsigned short f2bf(float f) {
    unsigned int u = __float_as_uint(f);
    u += 0x7FFFu + ((u >> 16) & 1u);           // round-to-nearest-even
    return (unsigned short)(u >> 16);
}

// A fragment (16x32 bf16, M x K): lane<16 -> row=lane, K {0..7, 16..23}
//                                  lane>=16 -> row=lane-16, K {8..15, 24..31}
__device__ __forceinline__ v16bf load_a_bf16(const unsigned short* rowp, int k0, int lane) {
    const int ofs = (lane & 16) ? 8 : 0;
    union { v16bf v; bf16x8 h[2]; } u;
    u.h[0] = *reinterpret_cast<const bf16x8*>(rowp + k0 + ofs);
    u.h[1] = *reinterpret_cast<const bf16x8*>(rowp + k0 + 16 + ofs);
    return u.v;
}

// A fragment from an fp32 row (probabilities), converted to bf16 on the fly.
__device__ __forceinline__ v16bf load_a_f32(const float* rowp, int k0, int lane) {
    const int ofs = (lane & 16) ? 8 : 0;
    f32x4 x0 = *reinterpret_cast<const f32x4*>(rowp + k0 + ofs);
    f32x4 x1 = *reinterpret_cast<const f32x4*>(rowp + k0 + ofs + 4);
    f32x4 x2 = *reinterpret_cast<const f32x4*>(rowp + k0 + 16 + ofs);
    f32x4 x3 = *reinterpret_cast<const f32x4*>(rowp + k0 + 16 + ofs + 4);
    union { v16bf v; unsigned short h[16]; } u;
#pragma unroll
    for (int i = 0; i < 4; ++i) {
        u.h[i]      = f2bf(x0[i]);
        u.h[4 + i]  = f2bf(x1[i]);
        u.h[8 + i]  = f2bf(x2[i]);
        u.h[12 + i] = f2bf(x3[i]);
    }
    return u.v;
}

// B fragment (32x16 bf16, K x N) out of the padded LDS tile.
// lane<16 -> col=lane, K {0..7, 16..23} ; lane>=16 -> col=lane-16, K {8..15, 24..31}
__device__ __forceinline__ v16bf load_b_lds(const unsigned short* ldsB, int j, int kk, int lane) {
    const int cj  = 16 * j + (lane & 15);
    const int ofs = (lane & 16) ? 8 : 0;
    const unsigned short* p = ldsB + cj * LDSB_PITCH + kk + ofs;
    union { v16bf v; bf16x8 h[2]; } u;
    u.h[0] = *reinterpret_cast<const bf16x8*>(p);
    u.h[1] = *reinterpret_cast<const bf16x8*>(p + 16);
    return u.v;
}

__device__ __forceinline__ v8f wmma_bf16(v16bf a, v16bf b, v8f c) {
    return __builtin_amdgcn_wmma_f32_16x16x32_bf16(false, a, false, b, (short)0, c, false, false);
}

// ---- Tensor Data Mover: DMA a 64(rows) x 64(bf16) tile, row stride `stride_elems`,
// ---- into LDS at byte offset `lds_off`, padding LDS rows to 72 halves (pad 4 dwords / 32 dwords).
__device__ __forceinline__ void tdm_load_tile(unsigned lds_off, const void* gptr, int stride_elems) {
    unsigned long long ga = (unsigned long long)(uintptr_t)gptr;
    u32x4 g0;
    g0[0] = 1u;                                                // count=1, user descriptor
    g0[1] = lds_off;                                           // lds_addr (bytes)
    g0[2] = (unsigned)(ga & 0xFFFFFFFFu);                      // global_addr[31:0]
    g0[3] = (unsigned)((ga >> 32) & 0x01FFFFFFu) | (2u << 30); // global_addr[56:32] | type=2
    i32x8 g1;
    g1[0] = (1 << 16)      // data_size = 2 bytes
          | (1 << 20)      // pad_enable
          | (4 << 22)      // pad_interval: pad after every 32 dwords (one 64-half row)
          | (3 << 25);     // pad_amount: 4 dwords (8 halves) -> 72-half pitch
    g1[1] = 64 << 16;      // tensor_dim0 = 64
    g1[2] = 64 << 16;      // tensor_dim1 = 64
    g1[3] = 64 << 16;      // tile_dim0   = 64
    g1[4] = 64;            // tile_dim1   = 64 (tile_dim2 = 0 -> 2D)
    g1[5] = stride_elems;  // tensor_dim0_stride[31:0]
    g1[6] = 0;
    g1[7] = 0;
    i32x4 g2 = {0, 0, 0, 0};                                   // D# group 2 (unused, 2D tile)
    i32x4 g3 = {0, 0, 0, 0};                                   // D# group 3 (unused, 2D tile)
    i32x8 g4 = {0, 0, 0, 0, 0, 0, 0, 0};                       // extra group (clang-23 6-arg form)
    __builtin_amdgcn_tensor_load_to_lds(g0, g1, g2, g3, g4, 0);
}

// one K-chunk (64 halves) of WMMA work, B from LDS, bf16 A from global.
// All four B fragments are fetched before the first WMMA so the scheduler can
// use partial DS-count waits instead of a full drain per matrix op.
__device__ __forceinline__ void mma_chunk_bf16(const unsigned short* arow, int k0, int lane,
                                               const unsigned short* ldsB, v8f acc[4]) {
#pragma unroll
    for (int kk = 0; kk < 64; kk += 32) {
        v16bf a  = load_a_bf16(arow, k0 + kk, lane);
        v16bf b0 = load_b_lds(ldsB, 0, kk, lane);
        v16bf b1 = load_b_lds(ldsB, 1, kk, lane);
        v16bf b2 = load_b_lds(ldsB, 2, kk, lane);
        v16bf b3 = load_b_lds(ldsB, 3, kk, lane);
        acc[0] = wmma_bf16(a, b0, acc[0]);
        acc[1] = wmma_bf16(a, b1, acc[1]);
        acc[2] = wmma_bf16(a, b2, acc[2]);
        acc[3] = wmma_bf16(a, b3, acc[3]);
    }
}

// same, fp32 A (probabilities) converted on the fly
__device__ __forceinline__ void mma_chunk_f32(const float* prow, int k0, int lane,
                                              const unsigned short* ldsB, v8f acc[4]) {
#pragma unroll
    for (int kk = 0; kk < 64; kk += 32) {
        v16bf a  = load_a_f32(prow, k0 + kk, lane);
        v16bf b0 = load_b_lds(ldsB, 0, kk, lane);
        v16bf b1 = load_b_lds(ldsB, 1, kk, lane);
        v16bf b2 = load_b_lds(ldsB, 2, kk, lane);
        v16bf b3 = load_b_lds(ldsB, 3, kk, lane);
        acc[0] = wmma_bf16(a, b0, acc[0]);
        acc[1] = wmma_bf16(a, b1, acc[1]);
        acc[2] = wmma_bf16(a, b2, acc[2]);
        acc[3] = wmma_bf16(a, b3, acc[3]);
    }
}

// ---------- kernel 1: fp32 -> bf16 convert (vectorized x4) ----------
__global__ void cvt_f32_bf16(const float* __restrict__ in, unsigned short* __restrict__ out, int n4) {
    int i = blockIdx.x * blockDim.x + threadIdx.x;
    if (i >= n4) return;
    f32x4 x = reinterpret_cast<const f32x4*>(in)[i];
    u16x4 o;
#pragma unroll
    for (int j = 0; j < 4; ++j) o[j] = f2bf(x[j]);
    reinterpret_cast<u16x4*>(out)[i] = o;
}

// ---------- kernel 2: QKV projection  Y = Xb @ Wb^T  (M=8192, N=3072, K=1024) ----------
__global__ void qkv_gemm(const unsigned short* __restrict__ xb,
                         const unsigned short* __restrict__ wb,
                         unsigned short* __restrict__ q,
                         unsigned short* __restrict__ k,
                         unsigned short* __restrict__ vt) {
    __shared__ unsigned short ldsB[2][64 * LDSB_PITCH];
    const int lane = threadIdx.x & 31;
    const int wv   = threadIdx.x >> 5;
    const int m0   = blockIdx.x * 128 + wv * 16;
    const int n0   = blockIdx.y * 64;

    const unsigned short* arow  = xb + (size_t)(m0 + (lane & 15)) * D_DIM;
    const unsigned short* bbase = wb + (size_t)n0 * D_DIM;   // 64 rows of W, stride 1024
    const unsigned lds0 = (unsigned)(uintptr_t)&ldsB[0][0];
    const unsigned lds1 = (unsigned)(uintptr_t)&ldsB[1][0];

    v8f acc[4];
#pragma unroll
    for (int j = 0; j < 4; ++j) acc[j] = {};

    const int NCH = D_DIM / 64;                              // 16 chunks
    if (threadIdx.x < 32) {
        tdm_load_tile(lds0, bbase, D_DIM);
        __builtin_amdgcn_s_wait_tensorcnt(0);
    }
    __syncthreads();
    for (int c = 0; c < NCH; ++c) {
        if (c + 1 < NCH && threadIdx.x < 32)                 // prefetch next chunk via TDM
            tdm_load_tile((c & 1) ? lds0 : lds1, bbase + (c + 1) * 64, D_DIM);
        __builtin_prefetch(arow + c * 64 + 128, 0, 1);
        mma_chunk_bf16(arow, c * 64, lane, ldsB[c & 1], acc);
        if (c + 1 < NCH) {
            if (threadIdx.x < 32) __builtin_amdgcn_s_wait_tensorcnt(0);
            __syncthreads();
        }
    }

    const int mB = m0 + ((lane & 16) ? 8 : 0);
    if (n0 < 1024) {                       // Q, fold softmax scale 1/sqrt(1024)
#pragma unroll
        for (int j = 0; j < 4; ++j) {
            const int n = n0 + 16 * j + (lane & 15);
#pragma unroll
            for (int i = 0; i < 8; ++i)
                q[(size_t)(mB + i) * D_DIM + n] = f2bf(acc[j][i] * 0.03125f);
        }
    } else if (n0 < 2048) {                // K
#pragma unroll
        for (int j = 0; j < 4; ++j) {
            const int n = n0 - 1024 + 16 * j + (lane & 15);
#pragma unroll
            for (int i = 0; i < 8; ++i)
                k[(size_t)(mB + i) * D_DIM + n] = f2bf(acc[j][i]);
        }
    } else {                               // V, stored transposed: vt[b][n][s]
        const int b  = m0 / S_LEN;
        const int s0 = (m0 % S_LEN) + ((lane & 16) ? 8 : 0);
#pragma unroll
        for (int j = 0; j < 4; ++j) {
            const int n = n0 - 2048 + 16 * j + (lane & 15);
            u16x8 o;
#pragma unroll
            for (int i = 0; i < 8; ++i) o[i] = f2bf(acc[j][i]);
            *reinterpret_cast<u16x8*>(vt + ((size_t)(b * D_DIM + n) * S_LEN + s0)) = o;
        }
    }
}

// ---------- kernel 3: scores = Q @ K^T (per batch; blocks above diagonal skipped) ----------
__global__ void scores_gemm(const unsigned short* __restrict__ q,
                            const unsigned short* __restrict__ k,
                            float* __restrict__ sc) {
    __shared__ unsigned short ldsB[2][64 * LDSB_PITCH];
    const int lane    = threadIdx.x & 31;
    const int wv      = threadIdx.x >> 5;
    const int b       = blockIdx.z;
    const int blockM0 = blockIdx.x * 128;
    const int qm0     = blockM0 + wv * 16;
    const int kn0     = blockIdx.y * 64;
    if (kn0 >= blockM0 + 128) return;      // block fully above diagonal (uniform exit)

    const unsigned short* arow  = q + (size_t)(b * S_LEN + qm0 + (lane & 15)) * D_DIM;
    const unsigned short* bbase = k + (size_t)(b * S_LEN + kn0) * D_DIM;  // 64 K-rows
    const unsigned lds0 = (unsigned)(uintptr_t)&ldsB[0][0];
    const unsigned lds1 = (unsigned)(uintptr_t)&ldsB[1][0];

    v8f acc[4];
#pragma unroll
    for (int j = 0; j < 4; ++j) acc[j] = {};

    const int NCH = D_DIM / 64;
    if (threadIdx.x < 32) {
        tdm_load_tile(lds0, bbase, D_DIM);
        __builtin_amdgcn_s_wait_tensorcnt(0);
    }
    __syncthreads();
    for (int c = 0; c < NCH; ++c) {
        if (c + 1 < NCH && threadIdx.x < 32)
            tdm_load_tile((c & 1) ? lds0 : lds1, bbase + (c + 1) * 64, D_DIM);
        __builtin_prefetch(arow + c * 64 + 128, 0, 1);
        mma_chunk_bf16(arow, c * 64, lane, ldsB[c & 1], acc);
        if (c + 1 < NCH) {
            if (threadIdx.x < 32) __builtin_amdgcn_s_wait_tensorcnt(0);
            __syncthreads();
        }
    }

    float* out = sc + (size_t)b * S_LEN * S_LEN;
    const int mB = qm0 + ((lane & 16) ? 8 : 0);
#pragma unroll
    for (int j = 0; j < 4; ++j) {
        const int n = kn0 + 16 * j + (lane & 15);
#pragma unroll
        for (int i = 0; i < 8; ++i)
            out[(size_t)(mB + i) * S_LEN + n] = acc[j][i];
    }
}

// ---------- kernel 4: causal softmax, one wave per row, in place ----------
__global__ void softmax_rows(float* __restrict__ sc) {
    const int lane = threadIdx.x & 31;
    const int wid  = blockIdx.x * (blockDim.x >> 5) + (threadIdx.x >> 5);
    const int qpos = wid & (S_LEN - 1);
    float* row = sc + (size_t)wid * S_LEN;

    float m = -3.402823466e38f;
    for (int c = lane; c <= qpos; c += 32) m = fmaxf(m, row[c]);
#pragma unroll
    for (int off = 16; off > 0; off >>= 1) m = fmaxf(m, __shfl_xor(m, off, 32));

    float s = 0.0f;
    for (int c = lane; c <= qpos; c += 32) s += __expf(row[c] - m);
#pragma unroll
    for (int off = 16; off > 0; off >>= 1) s += __shfl_xor(s, off, 32);
    const float inv = 1.0f / s;

    for (int c = lane; c < S_LEN; c += 32)
        row[c] = (c <= qpos) ? __expf(row[c] - m) * inv : 0.0f;   // zeros make PV dense
}

// ---------- kernel 5: O = P @ V (per batch; K-trip count truncated at block diagonal) ----------
__global__ void pv_gemm(const float* __restrict__ sc,
                        const unsigned short* __restrict__ vt,
                        float* __restrict__ out) {
    __shared__ unsigned short ldsB[2][64 * LDSB_PITCH];
    const int lane = threadIdx.x & 31;
    const int wv   = threadIdx.x >> 5;
    const int b    = blockIdx.z;
    const int qm0  = blockIdx.x * 128 + wv * 16;
    const int n0   = blockIdx.y * 64;

    const float* prow = sc + (size_t)(b * S_LEN + qm0 + (lane & 15)) * S_LEN;
    const unsigned short* bbase = vt + (size_t)(b * D_DIM + n0) * S_LEN; // 64 V^T rows, stride 2048
    const unsigned lds0 = (unsigned)(uintptr_t)&ldsB[0][0];
    const unsigned lds1 = (unsigned)(uintptr_t)&ldsB[1][0];

    v8f acc[4];
#pragma unroll
    for (int j = 0; j < 4; ++j) acc[j] = {};

    const int NCH = blockIdx.x * 2 + 2;    // keys beyond block diagonal are all zero
    if (threadIdx.x < 32) {
        tdm_load_tile(lds0, bbase, S_LEN);
        __builtin_amdgcn_s_wait_tensorcnt(0);
    }
    __syncthreads();
    for (int c = 0; c < NCH; ++c) {
        if (c + 1 < NCH && threadIdx.x < 32)
            tdm_load_tile((c & 1) ? lds0 : lds1, bbase + (c + 1) * 64, S_LEN);
        mma_chunk_f32(prow, c * 64, lane, ldsB[c & 1], acc);
        if (c + 1 < NCH) {
            if (threadIdx.x < 32) __builtin_amdgcn_s_wait_tensorcnt(0);
            __syncthreads();
        }
    }

    const int mB = qm0 + ((lane & 16) ? 8 : 0);
#pragma unroll
    for (int j = 0; j < 4; ++j) {
        const int n = n0 + 16 * j + (lane & 15);
#pragma unroll
        for (int i = 0; i < 8; ++i)
            out[(size_t)(b * S_LEN + mB + i) * D_DIM + n] = acc[j][i];
    }
}

// ---------- launch ----------
extern "C" void kernel_launch(void* const* d_in, const int* in_sizes, int n_in,
                              void* d_out, int out_size, void* d_ws, size_t ws_size,
                              hipStream_t stream) {
    const float* x  = (const float*)d_in[0];
    const float* wq = (const float*)d_in[1];
    const float* wk = (const float*)d_in[2];
    const float* wv = (const float*)d_in[3];
    float* outp = (float*)d_out;

    char* ws = (char*)d_ws;
    unsigned short* xb  = (unsigned short*)(ws);                       // 16 MB  [8192,1024] bf16
    unsigned short* wb  = (unsigned short*)(ws + (16u << 20));         //  6 MB  [3072,1024] bf16
    unsigned short* qb  = (unsigned short*)(ws + (24u << 20));         // 16 MB  [8192,1024] bf16
    unsigned short* kb  = (unsigned short*)(ws + (40u << 20));         // 16 MB  [8192,1024] bf16
    unsigned short* vtb = (unsigned short*)(ws + (56u << 20));         // 16 MB  [4,1024,2048] bf16 (V^T)
    float*          scb = (float*)(ws + (72u << 20));                  // 64 MB  [4,2048,2048] fp32

    // 1) fp32 -> bf16 conversions
    cvt_f32_bf16<<<8192, 256, 0, stream>>>(x,  xb, (BATCH * S_LEN * D_DIM) / 4);
    cvt_f32_bf16<<<1024, 256, 0, stream>>>(wq, wb,                     (D_DIM * D_DIM) / 4);
    cvt_f32_bf16<<<1024, 256, 0, stream>>>(wk, wb + 1 * D_DIM * D_DIM, (D_DIM * D_DIM) / 4);
    cvt_f32_bf16<<<1024, 256, 0, stream>>>(wv, wb + 2 * D_DIM * D_DIM, (D_DIM * D_DIM) / 4);

    // 2) fused QKV projection (Q scaled, V transposed)
    qkv_gemm<<<dim3(NROWS / 128, 3072 / 64, 1), 256, 0, stream>>>(xb, wb, qb, kb, vtb);

    // 3) causal scores
    scores_gemm<<<dim3(S_LEN / 128, S_LEN / 64, BATCH), 256, 0, stream>>>(qb, kb, scb);

    // 4) softmax rows (writes zeros above diagonal)
    softmax_rows<<<NROWS / 8, 256, 0, stream>>>(scb);

    // 5) P @ V
    pv_gemm<<<dim3(S_LEN / 128, D_DIM / 64, BATCH), 256, 0, stream>>>(scb, vtb, outp);
}
